// WindowAttention_63548336112005
// MI455X (gfx1250) — matrix operand order repaired and checked by reading, not compiled, not run on hardware
//
#include <hip/hip_runtime.h>

// ---------------------------------------------------------------------------
// Swin window attention, fused, bf16 WMMA (v_wmma_f32_16x16x32_bf16) on gfx1250
// ---------------------------------------------------------------------------

typedef __attribute__((ext_vector_type(16))) __bf16 v16bf;
typedef __attribute__((ext_vector_type(2)))  __bf16 v2bf;
typedef __attribute__((ext_vector_type(8)))  float  v8f;

union FragU { v16bf v; unsigned int u[8]; };

#define NTOK 64
#define CDIM 192
#define NHEADS 6
#define HD 32
#define XS_STRIDE 200   // 192 + 8 pad (bf16 elems)
#define QKV_STRIDE 40   // 32 + 8 pad
#define P_STRIDE 72     // 64 + 8 pad

// Packed f32x2 -> bf16x2. Prefer hardware cvt (builtin if declared, else the
// native __bf16 cast which lowers to v_cvt_[pk_]bf16_f32 when the target has
// it; worst case LLVM expands the same RNE sequence we'd write by hand).
__device__ __forceinline__ unsigned int pack2_bf16(float a, float b) {
#if __has_builtin(__builtin_amdgcn_cvt_pk_bf16_f32)
  v2bf r = __builtin_amdgcn_cvt_pk_bf16_f32(a, b);
  return __builtin_bit_cast(unsigned int, r);
#else
  v2bf r;
  r[0] = (__bf16)a;
  r[1] = (__bf16)b;
  return __builtin_bit_cast(unsigned int, r);
#endif
}

__device__ __forceinline__ unsigned short f32_to_bf16(float f) {
  __bf16 h = (__bf16)f;
  return __builtin_bit_cast(unsigned short, h);
}

__device__ __forceinline__ v8f wmma_bf16(v16bf a, v16bf b, v8f c) {
  // 8 args: (neg_a, A, neg_b, B, c_mod, C, reuse_a, reuse_b)
  return __builtin_amdgcn_wmma_f32_16x16x32_bf16(false, a, false, b, (short)0, c,
                                                 false, false);
}

// A fragment: A(M,K) = base[(row0+M)*stride + k0 + K], 16x32 bf16.
// Contiguous-K layout used here: lanes 0-15 rows M=0..15 (K 0..15), lanes 16-31 same
// rows (K 16..31).
__device__ __forceinline__ v16bf load_a(const unsigned short* base, int stride,
                                        int row0, int k0, int lane) {
  const int g = lane >> 4, m = lane & 15;
  const unsigned short* p = base + (row0 + m) * stride + k0 + g * 16;
  FragU f;
#pragma unroll
  for (int r = 0; r < 4; ++r)
    f.u[r] = *(const unsigned int*)(p + 2 * r);
#pragma unroll
  for (int r = 0; r < 4; ++r)
    f.u[4 + r] = *(const unsigned int*)(p + 8 + 2 * r);
  return f.v;
}

// B fragment where memory is [N][K] row-major: B(K,N) = base[(n0+N)*stride + k0 + K].
// Contiguous-K layout used here (32x16): lanes 0-15 K=0..15, lanes 16-31 K=16..31; VGPR r holds K=2r,2r+1.
__device__ __forceinline__ v16bf load_b_nk(const unsigned short* base, int stride,
                                           int n0, int k0, int lane) {
  const int g = lane >> 4, n = lane & 15;
  const unsigned short* p = base + (n0 + n) * stride + k0 + g * 16;
  FragU f;
#pragma unroll
  for (int r = 0; r < 8; ++r)
    f.u[r] = *(const unsigned int*)(p + 2 * r);
  return f.v;
}

// B fragment where memory is [K][N] row-major: B(K,N) = base[(k0+K)*stride + n0 + N].
__device__ __forceinline__ v16bf load_b_kn(const unsigned short* base, int stride,
                                           int k0, int n0, int lane) {
  const int g = lane >> 4, n = lane & 15;
  const unsigned short* p = base + (k0 + g * 16) * stride + n0 + n;
  FragU f;
#pragma unroll
  for (int r = 0; r < 8; ++r) {
    unsigned int lo = p[(2 * r) * stride];
    unsigned int hi = p[(2 * r + 1) * stride];
    f.u[r] = lo | (hi << 16);
  }
  return f.v;
}

// ---------------------------------------------------------------------------
// Prep: bf16-convert weights, pre-gather relative position bias -> [H][64][64]
// ---------------------------------------------------------------------------
__global__ void swin_prep_kernel(const float* __restrict__ qkv_w,
                                 const float* __restrict__ proj_w,
                                 const float* __restrict__ bias_table,
                                 const int* __restrict__ rel_index,
                                 unsigned short* __restrict__ wq,
                                 unsigned short* __restrict__ wp,
                                 float* __restrict__ wb) {
  int i = blockIdx.x * blockDim.x + threadIdx.x;
  // packed weight conversion: thread i handles elements 2i, 2i+1
  if (i < (3 * CDIM * CDIM) / 2) {
    const float2 w2 = ((const float2*)qkv_w)[i];
    ((unsigned int*)wq)[i] = pack2_bf16(w2.x, w2.y);
  }
  if (i < (CDIM * CDIM) / 2) {
    const float2 w2 = ((const float2*)proj_w)[i];
    ((unsigned int*)wp)[i] = pack2_bf16(w2.x, w2.y);
  }
  if (i < NHEADS * NTOK * NTOK) {
    int h = i / (NTOK * NTOK), nm = i % (NTOK * NTOK);
    wb[i] = bias_table[rel_index[nm] * NHEADS + h];
  }
}

// ---------------------------------------------------------------------------
// Fused window-attention kernel: one workgroup (8 waves) per window.
// ---------------------------------------------------------------------------
__global__ __launch_bounds__(256, 1) void swin_attn_kernel(
    const float* __restrict__ x, const float* __restrict__ mask,
    const float* __restrict__ qkv_b, const float* __restrict__ proj_b,
    const unsigned short* __restrict__ wq, const unsigned short* __restrict__ wp,
    const float* __restrict__ wb, float* __restrict__ out) {
  __shared__ __align__(16) unsigned short xs[NTOK * XS_STRIDE];            // x (bf16)
  __shared__ __align__(16) unsigned short qs[NHEADS * NTOK * QKV_STRIDE];  // q
  __shared__ __align__(16) unsigned short ks_[NHEADS * NTOK * QKV_STRIDE]; // k
  __shared__ __align__(16) unsigned short vs[NHEADS * NTOK * QKV_STRIDE];  // v
  __shared__ __align__(16) unsigned short ps[NHEADS * NTOK * P_STRIDE];    // softmax(P)
  __shared__ __align__(16) unsigned short os[NTOK * XS_STRIDE];            // attn out

  const int b = blockIdx.x;
  const int tid = threadIdx.x;
  const int wave = tid >> 5;
  const int lane = tid & 31;
  const int g = lane >> 4;   // half-wave
  const int ln = lane & 15;  // lane within half

  // ---- load x tile (float2), packed-convert to bf16 ----------------------
  const float2* xb2 = (const float2*)(x + (size_t)b * NTOK * CDIM);
  for (int i = tid; i < (NTOK * CDIM) / 2; i += 256) {
    int r = i / (CDIM / 2), c = i % (CDIM / 2);
    float2 xv = xb2[i];
    *(unsigned int*)&xs[r * XS_STRIDE + 2 * c] = pack2_bf16(xv.x, xv.y);
  }
  __syncthreads();

  // ---- Stage 1: qkv = x @ Wqkv^T + b  (64x576, K=192) --------------------
  for (int t = 0; t < 18; ++t) {
    int tile = wave + 8 * t;            // 0..143
    int mt = tile / 36, nt = tile % 36;
    v8f acc = {};
#pragma unroll
    for (int kb = 0; kb < 6; ++kb) {
      v16bf a = load_a(xs, XS_STRIDE, mt * 16, kb * 32, lane);
      v16bf bm = load_b_nk(wq, CDIM, nt * 16, kb * 32, lane);
      acc = wmma_bf16(a, bm, acc);
    }
    int o = nt * 16 + ln;               // output channel (per lane)
    float bias = qkv_b[o];
    int sid = o / CDIM;                 // 0=q 1=k 2=v (uniform per tile)
    int h = (o % CDIM) / HD;
    int d = o % HD;
    unsigned short* dst = (sid == 0) ? qs : (sid == 1) ? ks_ : vs;
    dst += h * NTOK * QKV_STRIDE + d;
#pragma unroll
    for (int v = 0; v < 8; v += 2) {
      int tok = mt * 16 + 8 * g + v;    // C-layout: M = 8*g + v
      unsigned int pk = pack2_bf16(acc[v] + bias, acc[v + 1] + bias);
      dst[tok * QKV_STRIDE]       = (unsigned short)(pk & 0xffffu);
      dst[(tok + 1) * QKV_STRIDE] = (unsigned short)(pk >> 16);
    }
  }
  __syncthreads();

  // ---- Stage 2a: S = scale*q@k^T + bias + mask ; softmax ; P -> LDS ------
  const float* maskb = mask + (size_t)b * NTOK * NTOK;
  const float scale = 5.656854249492380f;  // sqrt(head_dim)=sqrt(32)
  for (int t = 0; t < 3; ++t) {
    int rb = wave + 8 * t;              // 0..23 row-blocks
    int h = rb >> 2, i = rb & 3;
    const unsigned short* qb = qs + h * NTOK * QKV_STRIDE;
    const unsigned short* kb = ks_ + h * NTOK * QKV_STRIDE;
    v16bf a = load_a(qb, QKV_STRIDE, i * 16, 0, lane);
    float s[4][8];
#pragma unroll
    for (int j = 0; j < 4; ++j) {
      v16bf bm = load_b_nk(kb, QKV_STRIDE, j * 16, 0, lane);
      v8f acc = {};
      acc = wmma_bf16(a, bm, acc);
#pragma unroll
      for (int v = 0; v < 8; ++v) {
        int row = i * 16 + 8 * g + v;
        int col = j * 16 + ln;
        s[j][v] = acc[v] * scale + wb[h * 4096 + row * 64 + col] +
                  maskb[row * 64 + col];
      }
    }
    // row-wise softmax: row lives in one half-wave -> xor-shuffle widths 1..8
#pragma unroll
    for (int v = 0; v < 8; ++v) {
      float m = fmaxf(fmaxf(s[0][v], s[1][v]), fmaxf(s[2][v], s[3][v]));
#pragma unroll
      for (int off = 1; off < 16; off <<= 1)
        m = fmaxf(m, __shfl_xor(m, off, 32));
      float sum = 0.f;
#pragma unroll
      for (int j = 0; j < 4; ++j) { s[j][v] = __expf(s[j][v] - m); sum += s[j][v]; }
#pragma unroll
      for (int off = 1; off < 16; off <<= 1)
        sum += __shfl_xor(sum, off, 32);
      float inv = 1.0f / sum;
      int row = i * 16 + 8 * g + v;
      unsigned short* pdst = ps + h * NTOK * P_STRIDE + row * P_STRIDE + ln;
      unsigned int pk01 = pack2_bf16(s[0][v] * inv, s[1][v] * inv);
      unsigned int pk23 = pack2_bf16(s[2][v] * inv, s[3][v] * inv);
      pdst[0]  = (unsigned short)(pk01 & 0xffffu);
      pdst[16] = (unsigned short)(pk01 >> 16);
      pdst[32] = (unsigned short)(pk23 & 0xffffu);
      pdst[48] = (unsigned short)(pk23 >> 16);
    }
  }
  __syncthreads();

  // ---- Stage 2b: O = P @ V  (per head 64x32, K=64) -----------------------
  for (int t = 0; t < 6; ++t) {
    int tile = wave + 8 * t;            // 0..47
    int h = tile >> 3;
    int i = (tile >> 1) & 3;
    int dt = tile & 1;
    const unsigned short* pb = ps + h * NTOK * P_STRIDE;
    const unsigned short* vb = vs + h * NTOK * QKV_STRIDE;
    v8f acc = {};
#pragma unroll
    for (int kk = 0; kk < 2; ++kk) {
      v16bf a = load_a(pb, P_STRIDE, i * 16, kk * 32, lane);
      v16bf bm = load_b_kn(vb, QKV_STRIDE, kk * 32, dt * 16, lane);
      acc = wmma_bf16(a, bm, acc);
    }
#pragma unroll
    for (int v = 0; v < 8; v += 2) {
      int tok = i * 16 + 8 * g + v;
      unsigned int pk = pack2_bf16(acc[v], acc[v + 1]);
      os[tok * XS_STRIDE + h * HD + dt * 16 + ln]       = (unsigned short)(pk & 0xffffu);
      os[(tok + 1) * XS_STRIDE + h * HD + dt * 16 + ln] = (unsigned short)(pk >> 16);
    }
  }
  __syncthreads();

  // ---- Stage 3: out = O @ Wproj^T + b  (64x192, K=192) -------------------
  float* outb = out + (size_t)b * NTOK * CDIM;
  for (int t = 0; t < 6; ++t) {
    int tile = wave + 8 * t;            // 0..47
    int mt = tile / 12, nt = tile % 12;
    v8f acc = {};
#pragma unroll
    for (int kb = 0; kb < 6; ++kb) {
      v16bf a = load_a(os, XS_STRIDE, mt * 16, kb * 32, lane);
      v16bf bm = load_b_nk(wp, CDIM, nt * 16, kb * 32, lane);
      acc = wmma_bf16(a, bm, acc);
    }
    float pbv = proj_b[nt * 16 + ln];
#pragma unroll
    for (int v = 0; v < 8; ++v) {
      int tok = mt * 16 + 8 * g + v;
      outb[tok * CDIM + nt * 16 + ln] = acc[v] + pbv;
    }
  }
}

// ---------------------------------------------------------------------------
extern "C" void kernel_launch(void* const* d_in, const int* in_sizes, int n_in,
                              void* d_out, int out_size, void* d_ws, size_t ws_size,
                              hipStream_t stream) {
  const float* x          = (const float*)d_in[0];
  const float* mask       = (const float*)d_in[1];
  const float* qkv_w      = (const float*)d_in[2];
  const float* qkv_b      = (const float*)d_in[3];
  const float* proj_w     = (const float*)d_in[4];
  const float* proj_b     = (const float*)d_in[5];
  const float* bias_table = (const float*)d_in[6];
  const int*   rel_index  = (const int*)d_in[7];

  // workspace: bf16 qkv_w (576*192), bf16 proj_w (192*192), f32 bias [6][64][64]
  unsigned short* wq = (unsigned short*)d_ws;
  unsigned short* wp = wq + 3 * CDIM * CDIM;
  float*          wb = (float*)(wp + CDIM * CDIM);

  int prep_n = (3 * CDIM * CDIM) / 2;  // largest of the three prep ranges
  swin_prep_kernel<<<(prep_n + 255) / 256, 256, 0, stream>>>(
      qkv_w, proj_w, bias_table, rel_index, wq, wp, wb);

  swin_attn_kernel<<<4096, 256, 0, stream>>>(
      x, mask, qkv_b, proj_b, wq, wp, wb, (float*)d_out);
}
